// SelfAttentionModule_64811056497193
// MI455X (gfx1250) — matrix operand order repaired and checked by reading, not compiled, not run on hardware
//
#include <hip/hip_runtime.h>

// ---------------------------------------------------------------------------
// Self-attention (B=2, C=2048, N=4096) for gfx1250 / MI455X.
//  - all GEMMs: v_wmma_f32_16x16x32_bf16 (bf16 in, f32 accumulate)
//  - tile staging: global_load_async_to_lds_b128 (ASYNCcnt), 4-deep LDS
//    ring buffer, distance-2 prefetch, branch-free steady-state loop
//  - all operands pre-converted to bf16 once; every GEMM is D = A * B^T with
//    both operands row-major along K -> loaders are contiguous b128 copies
// ---------------------------------------------------------------------------

typedef __attribute__((ext_vector_type(16))) __bf16        v16bf;
typedef __attribute__((ext_vector_type(8)))  float         v8f;
typedef __attribute__((ext_vector_type(8)))  unsigned int  v8u;
typedef __attribute__((ext_vector_type(4)))  unsigned int  v4u;

// round-to-nearest-even f32 -> bf16
__device__ __forceinline__ unsigned short f2bf(float f) {
  unsigned u = __builtin_bit_cast(unsigned int, f);
  u += 0x7FFFu + ((u >> 16) & 1u);
  return (unsigned short)(u >> 16);
}

__device__ __forceinline__ v16bf combine8(v4u lo, v4u hi) {
  v8u u;
  u[0] = lo[0]; u[1] = lo[1]; u[2] = lo[2]; u[3] = lo[3];
  u[4] = hi[0]; u[5] = hi[1]; u[6] = hi[2]; u[7] = hi[3];
  return __builtin_bit_cast(v16bf, u);
}

// async memory -> LDS, 16 bytes per lane, tracked by ASYNCcnt
__device__ __forceinline__ void async_b128(unsigned lds_addr, const void* gaddr) {
  asm volatile("global_load_async_to_lds_b128 %0, %1, off"
               :: "v"(lds_addr), "v"(gaddr) : "memory");
}
__device__ __forceinline__ void wait_async8() {
  asm volatile("s_wait_asynccnt 0x8" ::: "memory");
}
__device__ __forceinline__ void wait_async4() {
  asm volatile("s_wait_asynccnt 0x4" ::: "memory");
}
__device__ __forceinline__ void wait_async0() {
  asm volatile("s_wait_asynccnt 0x0" ::: "memory");
}

#define LDSS 40                  // LDS row stride in halfwords (80 B = 16*5)
#define TILE_H (128 * LDSS)      // halfwords per 128x32 tile buffer
#define TILE_BYTES (TILE_H * 2)  // 10240 B
#define NBUF 4                   // ring depth (distance-2 prefetch)

// D[M,N'] = alpha * A[M,K] * B[N',K]^T (+ bias[m])
// A, B bf16 row-major along K.  OUT_BF16: D stored bf16, else f32.
// DT: store transposed D[n][m] (only used with OUT_BF16).
template <bool DT, bool HAS_BIAS, bool OUT_BF16>
__global__ __launch_bounds__(256)
void gemm_wmma_async(const unsigned short* __restrict__ A,
                     const unsigned short* __restrict__ Bm,
                     const float* __restrict__ bias,
                     void* __restrict__ Dout,
                     int Kdim, int lda, int ldb, int ldd,
                     long long sAb, long long sBb, long long sDb,
                     float alpha)
{
  __shared__ __align__(16) unsigned short As[NBUF * TILE_H];
  __shared__ __align__(16) unsigned short Bs[NBUF * TILE_H];

  A  += (long long)blockIdx.z * sAb;
  Bm += (long long)blockIdx.z * sBb;

  const int mBlk = blockIdx.y * 128;
  const int nBlk = blockIdx.x * 128;

  const int tid  = threadIdx.x;
  const int lane = tid & 31;
  const int wave = tid >> 5;
  const int wm   = wave >> 1;   // 0..3 -> 32-row band of M
  const int wn   = wave & 1;    // 0..1 -> 64-col band of N'
  const int g    = lane >> 4;
  const int l15  = lane & 15;

  // per-thread staging slots: 2 b128 loads for A, 2 for B per 128x32 tile
  const unsigned short* gA[2]; const unsigned short* gB[2];
  unsigned lA[2], lB[2];
#pragma unroll
  for (int j = 0; j < 2; ++j) {
    int idx = tid + j * 256;           // 0..511
    int row = idx >> 2;                // 0..127
    int kb  = (idx & 3) << 3;          // 0,8,16,24
    gA[j] = A  + (long long)(mBlk + row) * lda + kb;
    gB[j] = Bm + (long long)(nBlk + row) * ldb + kb;
    lA[j] = (unsigned)(size_t)&As[row * LDSS + kb];
    lB[j] = (unsigned)(size_t)&Bs[row * LDSS + kb];
  }

  v8f acc[2][4];
#pragma unroll
  for (int mt = 0; mt < 2; ++mt)
#pragma unroll
    for (int nt = 0; nt < 4; ++nt)
      acc[mt][nt] = (v8f){0.f,0.f,0.f,0.f,0.f,0.f,0.f,0.f};

  // issue one 128x32 A tile + one 128x32 B tile into ring slot `buf`
  auto issue_tile = [&](unsigned buf) {
    const unsigned off = buf * TILE_BYTES;
#pragma unroll
    for (int j = 0; j < 2; ++j) {
      async_b128(lA[j] + off, gA[j]); gA[j] += 32;
      async_b128(lB[j] + off, gB[j]); gB[j] += 32;
    }
  };

  // consume ring slot `buf`: fragment loads + 8 WMMAs
  auto compute_tile = [&](int buf) {
    const unsigned short* Asb = As + buf * TILE_H;
    const unsigned short* Bsb = Bs + buf * TILE_H;
    v16bf af[2], bfm[4];
#pragma unroll
    for (int mt = 0; mt < 2; ++mt) {
      const unsigned short* rp = &Asb[(wm * 32 + mt * 16 + l15) * LDSS];
      v4u lo = *reinterpret_cast<const v4u*>(rp + (g << 3));       // K = g*8+0..7
      v4u hi = *reinterpret_cast<const v4u*>(rp + 16 + (g << 3));  // K = 16+g*8+0..7
      af[mt] = combine8(lo, hi);
    }
#pragma unroll
    for (int nt = 0; nt < 4; ++nt) {
      const unsigned short* rp = &Bsb[(wn * 64 + nt * 16 + l15) * LDSS];
      v4u lo = *reinterpret_cast<const v4u*>(rp + (g << 3));       // K = g*8+0..7
      v4u hi = *reinterpret_cast<const v4u*>(rp + 16 + (g << 3));  // K = 16+g*8+0..7
      bfm[nt] = combine8(lo, hi);
    }
#pragma unroll
    for (int mt = 0; mt < 2; ++mt)
#pragma unroll
      for (int nt = 0; nt < 4; ++nt)
        acc[mt][nt] = __builtin_amdgcn_wmma_f32_16x16x32_bf16(
            false, af[mt], false, bfm[nt], (short)0, acc[mt][nt], false, false);
  };

  const int T = Kdim >> 5;             // 64 or 128 (always >= 2)

  // pipeline prologue: tiles 0 and 1 in flight
  issue_tile(0);
  issue_tile(1);

  // steady state: branch-free; tile it+2 in flight while computing tile it
  for (int it = 0; it < T - 2; ++it) {
    issue_tile((unsigned)(it + 2) & (NBUF - 1));
    wait_async8();                     // tile `it` landed (2 tiles still flying)
    __syncthreads();
    compute_tile(it & (NBUF - 1));
    __syncthreads();                   // slot reusable at iteration it+NBUF-2
  }

  // epilogue: last two tiles
  wait_async4();
  __syncthreads();
  compute_tile((T - 2) & (NBUF - 1));
  __syncthreads();
  wait_async0();
  __syncthreads();
  compute_tile((T - 1) & (NBUF - 1));

  // ---- store D (C/D layout: vgpr r -> M = g*8 + r, N = lane&15) ----
#pragma unroll
  for (int mt = 0; mt < 2; ++mt)
#pragma unroll
    for (int nt = 0; nt < 4; ++nt) {
      const int mBase = mBlk + wm * 32 + mt * 16 + g * 8;
      const int nIdx  = nBlk + wn * 64 + nt * 16 + l15;
      float vals[8];
#pragma unroll
      for (int r = 0; r < 8; ++r) {
        vals[r] = acc[mt][nt][r] * alpha;
        if (HAS_BIAS) vals[r] += bias[mBase + r];
      }
      if (OUT_BF16) {
        unsigned short* D = (unsigned short*)Dout + (long long)blockIdx.z * sDb;
        if (DT) {                      // D[n][m], 8 consecutive m -> one b128
          v4u pk;
#pragma unroll
          for (int r2 = 0; r2 < 4; ++r2)
            pk[r2] = (unsigned)f2bf(vals[2 * r2]) |
                     ((unsigned)f2bf(vals[2 * r2 + 1]) << 16);
          *reinterpret_cast<v4u*>(D + (long long)nIdx * ldd + mBase) = pk;
        } else {
#pragma unroll
          for (int r = 0; r < 8; ++r)
            D[(long long)(mBase + r) * ldd + nIdx] = f2bf(vals[r]);
        }
      } else {
        float* D = (float*)Dout + (long long)blockIdx.z * sDb;
#pragma unroll
        for (int r = 0; r < 8; ++r)
          D[(long long)(mBase + r) * ldd + nIdx] = vals[r];
      }
    }
}

// xT[b][n][c] (bf16) = concat(f_rgb,f_i)[b][c][n], tiled 32x32 transpose
__global__ __launch_bounds__(256)
void transpose_x_bf16(const float* __restrict__ f_rgb, const float* __restrict__ f_i,
                      unsigned short* __restrict__ xT)
{
  __shared__ unsigned short tile[32][33];
  const long long N = 4096, C = 2048;
  const int b = blockIdx.z;
  const int nBase = blockIdx.x * 32;
  const int cBase = blockIdx.y * 32;        // 0..4064, never straddles C
  const int tx = threadIdx.x & 31;
  const int ty = threadIdx.x >> 5;          // 0..7
  const float* src = (cBase < (int)C)
      ? (f_rgb + ((long long)b * C + cBase) * N)
      : (f_i   + ((long long)b * C + (cBase - (int)C)) * N);
#pragma unroll
  for (int i = 0; i < 4; ++i) {
    int r = ty + i * 8;
    tile[r][tx] = f2bf(src[(long long)r * N + nBase + tx]);
  }
  __syncthreads();
  unsigned short* dst = xT + ((long long)b * N + nBase) * (2 * C) + cBase;
#pragma unroll
  for (int i = 0; i < 4; ++i) {
    int r = ty + i * 8;
    dst[(long long)r * (2 * C) + tx] = tile[tx][r];
  }
}

// bf16 weight copies; wvs = wv[:2048]+wv[2048:], bvs = bv[:2048]+bv[2048:]
__global__ __launch_bounds__(256)
void conv_weights(const float* __restrict__ wq, const float* __restrict__ wk,
                  const float* __restrict__ wv, const float* __restrict__ bv,
                  unsigned short* __restrict__ wqb, unsigned short* __restrict__ wkb,
                  unsigned short* __restrict__ wvsb, float* __restrict__ bvs)
{
  long long i = (long long)blockIdx.x * 256 + threadIdx.x;
  const long long half = 2048LL * 4096LL;
  if (i < half) {
    wqb[i]  = f2bf(wq[i]);
    wkb[i]  = f2bf(wk[i]);
    wvsb[i] = f2bf(wv[i] + wv[i + half]);
  }
  if (i < 2048) bvs[i] = bv[i] + bv[i + 2048];
}

// in-place softmax over 4096-wide rows; also emits bf16 shadow copy
__global__ __launch_bounds__(256)
void softmax_rows(float* __restrict__ attn, unsigned short* __restrict__ attn_bf)
{
  const long long rowOff = (long long)blockIdx.x * 4096;
  float* row = attn + rowOff;
  unsigned short* rowb = attn_bf + rowOff;
  const int t = threadIdx.x;
  float vals[16];
  float m = -3.4028235e38f;
#pragma unroll
  for (int i = 0; i < 16; ++i) { vals[i] = row[t + i * 256]; m = fmaxf(m, vals[i]); }

  __shared__ float red[256];
  red[t] = m; __syncthreads();
  for (int s = 128; s > 0; s >>= 1) {
    if (t < s) red[t] = fmaxf(red[t], red[t + s]);
    __syncthreads();
  }
  m = red[0]; __syncthreads();

  float sum = 0.f;
#pragma unroll
  for (int i = 0; i < 16; ++i) { vals[i] = __expf(vals[i] - m); sum += vals[i]; }
  red[t] = sum; __syncthreads();
  for (int s = 128; s > 0; s >>= 1) {
    if (t < s) red[t] += red[t + s];
    __syncthreads();
  }
  const float inv = 1.f / red[0];
#pragma unroll
  for (int i = 0; i < 16; ++i) {
    float o = vals[i] * inv;
    row[t + i * 256]  = o;
    rowb[t + i * 256] = f2bf(o);
  }
}

extern "C" void kernel_launch(void* const* d_in, const int* in_sizes, int n_in,
                              void* d_out, int out_size, void* d_ws, size_t ws_size,
                              hipStream_t stream)
{
  (void)in_sizes; (void)n_in; (void)out_size; (void)ws_size;

  const float* f_rgb = (const float*)d_in[0];
  const float* f_i   = (const float*)d_in[1];
  const float* wq    = (const float*)d_in[2];
  const float* bq    = (const float*)d_in[3];
  const float* wk    = (const float*)d_in[4];
  const float* bk    = (const float*)d_in[5];
  const float* wv    = (const float*)d_in[6];
  const float* bv    = (const float*)d_in[7];

  const long long C = 2048, N = 4096, Bn = 2;
  const long long img = C * N;                 // per-batch tensor elems

  float* out_final = (float*)d_out;            // [B, C, N] f32
  float* out_rgb   = out_final + Bn * img;
  float* out_fi    = out_rgb   + Bn * img;
  float* out_attn  = out_fi    + Bn * img;     // [B, N, N] f32

  // bf16 workspace
  unsigned short* ws16    = (unsigned short*)d_ws;
  unsigned short* xT      = ws16;                       // [B, N, 2C]
  unsigned short* Qt      = xT      + Bn * N * 2 * C;   // [B, N, C]
  unsigned short* Kt      = Qt      + Bn * N * C;       // [B, N, C]
  unsigned short* Vp      = Kt      + Bn * N * C;       // [B, C, N]
  unsigned short* attn_bf = Vp      + Bn * C * N;       // [B, N, N]
  unsigned short* wqb     = attn_bf + Bn * N * N;       // [C, 2C]
  unsigned short* wkb     = wqb     + C * 2 * C;
  unsigned short* wvsb    = wkb     + C * 2 * C;
  float*          bvs     = (float*)(wvsb + C * 2 * C); // [C]

  // --- preprocessing: one-time f32 -> bf16 conversion / layout change ---
  transpose_x_bf16<<<dim3(128, 128, 2), dim3(256), 0, stream>>>(f_rgb, f_i, xT);
  conv_weights<<<dim3(32768), dim3(256), 0, stream>>>(wq, wk, wv, bv,
                                                      wqb, wkb, wvsb, bvs);

  // Qt[N,C] = (wq @ x + bq)^T  : A=wqb[C,2C], B=xT[N,2C]
  gemm_wmma_async<true, true, true><<<dim3(32, 16, 2), dim3(256), 0, stream>>>(
      wqb, xT, bq, Qt, (int)(2 * C), (int)(2 * C), (int)(2 * C), (int)C,
      0LL, N * 2 * C, N * C, 1.0f);

  // Kt[N,C] = (wk @ x + bk)^T
  gemm_wmma_async<true, true, true><<<dim3(32, 16, 2), dim3(256), 0, stream>>>(
      wkb, xT, bk, Kt, (int)(2 * C), (int)(2 * C), (int)(2 * C), (int)C,
      0LL, N * 2 * C, N * C, 1.0f);

  // Vp[C,N] = (wv_top+wv_bot) @ x + (bv_top+bv_bot)
  gemm_wmma_async<false, true, true><<<dim3(32, 16, 2), dim3(256), 0, stream>>>(
      wvsb, xT, bvs, Vp, (int)(2 * C), (int)(2 * C), (int)(2 * C), (int)N,
      0LL, N * 2 * C, C * N, 1.0f);

  // scores (f32, pre-softmax): attn = Qt @ Kt^T / 64
  gemm_wmma_async<false, false, false><<<dim3(32, 32, 2), dim3(256), 0, stream>>>(
      Qt, Kt, nullptr, out_attn, (int)C, (int)C, (int)C, (int)N,
      N * C, N * C, N * N, 1.0f / 64.0f);

  softmax_rows<<<dim3(8192), dim3(256), 0, stream>>>(out_attn, attn_bf);

  // f_final[C,N] = Vp @ attn_bf^T
  gemm_wmma_async<false, false, false><<<dim3(32, 16, 2), dim3(256), 0, stream>>>(
      Vp, attn_bf, nullptr, out_final, (int)N, (int)N, (int)N, (int)N,
      C * N, N * N, C * N, 1.0f);

  hipMemcpyAsync(out_rgb, f_rgb, (size_t)(Bn * img * sizeof(float)),
                 hipMemcpyDeviceToDevice, stream);
  hipMemcpyAsync(out_fi, f_i, (size_t)(Bn * img * sizeof(float)),
                 hipMemcpyDeviceToDevice, stream);
}
